// CategoricalGINEncoder_67044439491024
// MI455X (gfx1250) — matrix operand order repaired and checked by reading, not compiled, not run on hardware
//
#include <hip/hip_runtime.h>
#include <hip/hip_bf16.h>

// CDNA5 / gfx1250: wave32, WMMA 16x16x32 bf16 -> f32 accumulate.
// Workspace layout (bytes), feat = N*256:
//   [0,          feat*4)   bufX   f32 node features (x, later LN1 output h)
//   [feat*4,     feat*8)   bufAgg f32 scatter accumulator (later GEMM-b f32 out)
//   [feat*8,     feat*10)  bufA   bf16 GEMM A matrix (x+agg)
//   [feat*10,    feat*12)  bufH1  bf16 relu(GEMM-a) output
//   [feat*12, +512KB)      Wbf    bf16 transposed weights, 4 x 256*256
// Total ~= 307.7 MB for N=100000.

typedef __attribute__((ext_vector_type(8)))  float  v8f;
typedef __attribute__((ext_vector_type(8)))  __bf16 v8bf;
typedef __attribute__((ext_vector_type(16))) __bf16 v16bf;

#define LN_EPS 1e-5f

// LDS B-slab: 64 columns x 256 K, padded to 264 per column so that the
// column stride is 528 B = 132 dwords; 132 mod 64 = 4 -> 16 lanes reading
// b128 at stride 528 B cover banks 0,4,...,60 with no conflicts.
#define BCOL_PAD 264

__device__ __forceinline__ v16bf ld_frag16(const __bf16* p0, const __bf16* p1) {
  v8bf lo = *(const v8bf*)p0;
  v8bf hi = *(const v8bf*)p1;
  return __builtin_shufflevector(lo, hi, 0, 1, 2, 3, 4, 5, 6, 7,
                                 8, 9, 10, 11, 12, 13, 14, 15);
}

// ---------------- embedding gather: x[node, field*64 + j] = emb_field[idx][j]
__global__ void embed_kernel(const int* __restrict__ xcat,
                             const float* __restrict__ e0, const float* __restrict__ e1,
                             const float* __restrict__ e2, const float* __restrict__ e3,
                             float* __restrict__ x, int n) {
  int t = blockIdx.x * blockDim.x + threadIdx.x;
  if (t >= n * 256) return;
  int node  = t >> 8;
  int col   = t & 255;
  int field = col >> 6;
  int j     = col & 63;
  int idx   = xcat[node * 4 + field];
  const float* tab = (field == 0) ? e0 : (field == 1) ? e1 : (field == 2) ? e2 : e3;
  x[t] = tab[(size_t)idx * 64 + j];
}

// ---------------- weight transpose + f32->bf16: Wt[m][n*256 + k] = W_m[k*256 + n]
__global__ void wcvt_kernel(const float* __restrict__ w0, const float* __restrict__ w1,
                            const float* __restrict__ w2, const float* __restrict__ w3,
                            __bf16* __restrict__ out) {
  int t = blockIdx.x * blockDim.x + threadIdx.x;   // t < 4*65536
  int m = t >> 16;
  int r = t & 65535;
  int k = r >> 8;
  int n = r & 255;
  const float* w = (m == 0) ? w0 : (m == 1) ? w1 : (m == 2) ? w2 : w3;
  out[(size_t)m * 65536 + (size_t)n * 256 + k] = (__bf16)w[(size_t)k * 256 + n];
}

// ---------------- edge scatter-add: agg[dst] += x[src], 4 floats per thread
__global__ void scatter_kernel(const float* __restrict__ x,
                               const int* __restrict__ src, const int* __restrict__ dst,
                               float* __restrict__ agg, long long total) {
  long long t = (long long)blockIdx.x * blockDim.x + threadIdx.x;
  if (t >= total) return;
  int e = (int)(t >> 6);
  int c = (int)(t & 63);
  int s = src[e];
  int d = dst[e];
  const float4 v = ((const float4*)(x + (size_t)s * 256))[c];
  float* a = agg + (size_t)d * 256 + c * 4;
  __hip_atomic_fetch_add(a + 0, v.x, __ATOMIC_RELAXED, __HIP_MEMORY_SCOPE_AGENT);
  __hip_atomic_fetch_add(a + 1, v.y, __ATOMIC_RELAXED, __HIP_MEMORY_SCOPE_AGENT);
  __hip_atomic_fetch_add(a + 2, v.z, __ATOMIC_RELAXED, __HIP_MEMORY_SCOPE_AGENT);
  __hip_atomic_fetch_add(a + 3, v.w, __ATOMIC_RELAXED, __HIP_MEMORY_SCOPE_AGENT);
}

// ---------------- (x + agg) -> bf16 A matrix
__global__ void add_cvt_kernel(const float* __restrict__ x, const float* __restrict__ agg,
                               __bf16* __restrict__ a, long long n) {
  long long t = (long long)blockIdx.x * blockDim.x + threadIdx.x;
  if (t >= n) return;
  a[t] = (__bf16)(x[t] + agg[t]);
}

// ---------------- WMMA GEMM: out[M,256] = act(A[M,256] @ W[256,256] + bias)
// A: bf16 row-major. Wt: bf16 column-major (Wt[n*256+k]).
// Block: 128 threads = 4 waves; block tile 128(M) x 64(N).
// Weights for the 64-column slab staged in LDS once per block; each wave
// owns 32 rows -> 2 M-subtiles x 4 N-tiles = 8 f32 accumulators.
// Software-pipelined K loop: fragments for step k+1 (2 A from global,
// 4 B from LDS) are issued into a second register set before the 8 WMMAs
// of step k, so DS/L2 latency overlaps the matrix pipe.
// CDNA5 16-bit A layout: lane holds M = lane&15; its 16 bf16 are
// K in {kb..kb+7} U {kb+16..kb+23}, kb = (lane>>4)*8.
// B layout: lane holds N = lane&15, K in the same order as A.
// C/D layout: VGPR r holds (M = r + 8*(lane>>4), N = lane&15).
template <bool RELU, typename OutT>
__global__ void gemm_wmma(const __bf16* __restrict__ A, const __bf16* __restrict__ Wt,
                          const float* __restrict__ bias, OutT* __restrict__ out, int M) {
  __shared__ __bf16 lB[64 * BCOL_PAD];    // ~33 KB

  const int lane    = threadIdx.x & 31;
  const int wave    = threadIdx.x >> 5;
  const int colBase = blockIdx.y * 64;

  // Cooperative stage of the 64-column weight slab (64 x 256 bf16 = 32 KB).
  {
    const float4* gsrc = (const float4*)(Wt + (size_t)colBase * 256);
    for (int i = threadIdx.x; i < 64 * 32; i += blockDim.x) {
      int col = i >> 5;           // 0..63
      int j   = i & 31;           // 16B chunk within the 512B column
      *(float4*)(lB + col * BCOL_PAD + j * 8) = gsrc[col * 32 + j];
    }
  }
  __syncthreads();                // only barrier; uniform exits below are safe

  const int rowBase = blockIdx.x * 128 + wave * 32;
  if (rowBase >= M) return;       // wave-uniform: EXEC stays all-ones

  const int l15  = lane & 15;
  const int half = lane >> 4;

  const __bf16* aRow0 = A + (size_t)(rowBase + l15) * 256 + half * 8;
  const __bf16* aRow1 = A + (size_t)(rowBase + 16 + l15) * 256 + half * 8;
  const __bf16* bRow  = lB + (size_t)l15 * BCOL_PAD + half * 8;

  v8f acc[4][2] = {};
  v16bf a0, a1, bf[4];

  // prologue: fragments for k0 = 0
  a0 = ld_frag16(aRow0, aRow0 + 16);
  a1 = ld_frag16(aRow1, aRow1 + 16);
#pragma unroll
  for (int t = 0; t < 4; ++t)
    bf[t] = ld_frag16(bRow + t * 16 * BCOL_PAD, bRow + t * 16 * BCOL_PAD + 16);

#pragma unroll
  for (int k0 = 0; k0 < 256; k0 += 32) {
    v16bf na0, na1, nbf[4];
    if (k0 + 32 < 256) {
      const __bf16* ap0 = aRow0 + (k0 + 32);
      const __bf16* ap1 = aRow1 + (k0 + 32);
      na0 = ld_frag16(ap0, ap0 + 16);
      na1 = ld_frag16(ap1, ap1 + 16);
#pragma unroll
      for (int t = 0; t < 4; ++t) {
        const __bf16* bp = bRow + t * 16 * BCOL_PAD + (k0 + 32);
        nbf[t] = ld_frag16(bp, bp + 16);
      }
    }
#pragma unroll
    for (int t = 0; t < 4; ++t) {
      acc[t][0] = __builtin_amdgcn_wmma_f32_16x16x32_bf16(
          false, a0, false, bf[t], (short)0, acc[t][0], false, false);
      acc[t][1] = __builtin_amdgcn_wmma_f32_16x16x32_bf16(
          false, a1, false, bf[t], (short)0, acc[t][1], false, false);
    }
    if (k0 + 32 < 256) {
      a0 = na0;
      a1 = na1;
#pragma unroll
      for (int t = 0; t < 4; ++t) bf[t] = nbf[t];
    }
  }

#pragma unroll
  for (int t = 0; t < 4; ++t) {
    int col  = colBase + t * 16 + l15;
    float bv = bias[col];
#pragma unroll
    for (int m = 0; m < 2; ++m) {
#pragma unroll
      for (int r = 0; r < 8; ++r) {
        int row = rowBase + m * 16 + r + half * 8;
        if (row < M) {
          float v = acc[t][m][r] + bv;
          if (RELU) v = v > 0.f ? v : 0.f;
          out[(size_t)row * 256 + col] = (OutT)v;
        }
      }
    }
  }
}

// ---------------- per-row LayerNorm (256 cols), optional fused relu on input
template <bool RELU>
__global__ void ln_kernel(const float* __restrict__ in, const float* __restrict__ g,
                          const float* __restrict__ b, float* __restrict__ out) {
  __shared__ float s1[256];
  __shared__ float s2[256];
  int row = blockIdx.x;
  int c   = threadIdx.x;
  float v = in[(size_t)row * 256 + c];
  if (RELU) v = fmaxf(v, 0.f);
  s1[c] = v;
  s2[c] = v * v;
  __syncthreads();
#pragma unroll
  for (int off = 128; off > 0; off >>= 1) {
    if (c < off) {
      s1[c] += s1[c + off];
      s2[c] += s2[c + off];
    }
    __syncthreads();
  }
  float m   = s1[0] * (1.0f / 256.0f);
  float var = s2[0] * (1.0f / 256.0f) - m * m;
  float inv = rsqrtf(var + LN_EPS);
  out[(size_t)row * 256 + c] = (v - m) * inv * g[c] + b[c];
}

extern "C" void kernel_launch(void* const* d_in, const int* in_sizes, int n_in,
                              void* d_out, int out_size, void* d_ws, size_t ws_size,
                              hipStream_t stream) {
  const int*   xcat = (const int*)d_in[0];
  const int*   eidx = (const int*)d_in[1];
  const float* e0   = (const float*)d_in[2];
  const float* e1   = (const float*)d_in[3];
  const float* e2   = (const float*)d_in[4];
  const float* e3   = (const float*)d_in[5];
  const float* w1a  = (const float*)d_in[6];
  const float* b1a  = (const float*)d_in[7];
  const float* w1b  = (const float*)d_in[8];
  const float* b1b  = (const float*)d_in[9];
  const float* w2a  = (const float*)d_in[10];
  const float* b2a  = (const float*)d_in[11];
  const float* w2b  = (const float*)d_in[12];
  const float* b2b  = (const float*)d_in[13];
  const float* ln1g = (const float*)d_in[14];
  const float* ln1b = (const float*)d_in[15];
  const float* ln2g = (const float*)d_in[16];
  const float* ln2b = (const float*)d_in[17];

  const int N = in_sizes[0] / 4;   // 100000 nodes
  const int E = in_sizes[1] / 2;   // 1.6M edges
  const long long feat = (long long)N * 256;

  char*   ws     = (char*)d_ws;
  float*  bufX   = (float*)ws;
  float*  bufAgg = (float*)(ws + (size_t)feat * 4);
  __bf16* bufA   = (__bf16*)(ws + (size_t)feat * 8);
  __bf16* bufH1  = (__bf16*)(ws + (size_t)feat * 10);
  __bf16* Wbf    = (__bf16*)(ws + (size_t)feat * 12);

  const int* src = eidx;
  const int* dst = eidx + E;

  const int   threads   = 256;
  const int   featBlks  = (int)((feat + threads - 1) / threads);
  const long long stot  = (long long)E * 64;
  const int   scatBlks  = (int)((stot + threads - 1) / threads);
  dim3 gemmGrid((N + 127) / 128, 4);
  dim3 gemmBlk(128);

  // weights -> bf16 transposed (once per launch; deterministic)
  wcvt_kernel<<<(4 * 65536) / threads, threads, 0, stream>>>(w1a, w1b, w2a, w2b, Wbf);
  // embedding gather
  embed_kernel<<<featBlks, threads, 0, stream>>>(xcat, e0, e1, e2, e3, bufX, N);

  // ---- GIN conv 1 ----
  hipMemsetAsync(bufAgg, 0, (size_t)feat * 4, stream);
  scatter_kernel<<<scatBlks, threads, 0, stream>>>(bufX, src, dst, bufAgg, stot);
  add_cvt_kernel<<<featBlks, threads, 0, stream>>>(bufX, bufAgg, bufA, feat);
  gemm_wmma<true,  __bf16><<<gemmGrid, gemmBlk, 0, stream>>>(bufA, Wbf,          b1a, bufH1, N);
  gemm_wmma<false, float ><<<gemmGrid, gemmBlk, 0, stream>>>(bufH1, Wbf + 65536, b1b, bufAgg, N);
  // relu + LayerNorm 1 -> new node features in bufX
  ln_kernel<true><<<N, 256, 0, stream>>>(bufAgg, ln1g, ln1b, bufX);

  // ---- GIN conv 2 ----
  hipMemsetAsync(bufAgg, 0, (size_t)feat * 4, stream);
  scatter_kernel<<<scatBlks, threads, 0, stream>>>(bufX, src, dst, bufAgg, stot);
  add_cvt_kernel<<<featBlks, threads, 0, stream>>>(bufX, bufAgg, bufA, feat);
  gemm_wmma<true,  __bf16><<<gemmGrid, gemmBlk, 0, stream>>>(bufA, Wbf + 2 * 65536, b2a, bufH1, N);
  gemm_wmma<false, float ><<<gemmGrid, gemmBlk, 0, stream>>>(bufH1, Wbf + 3 * 65536, b2b, bufAgg, N);
  // LayerNorm 2 -> final output
  ln_kernel<false><<<N, 256, 0, stream>>>(bufAgg, ln2g, ln2b, (float*)d_out);
}
